// RewaHierarchicalAttention_90237262889099
// MI455X (gfx1250) — matrix operand order repaired and checked by reading, not compiled, not run on hardware
//
#include <hip/hip_runtime.h>
#include <hip/hip_bf16.h>
#include <math.h>

// ---- problem constants (match reference) ----
#define BB   2
#define NN   4096
#define EE   1024
#define HH   16
#define HD   64
#define MN   (BB * NN)          // 8192 rows
#define RR   8
#define LORA_SCALE 2.0f

typedef _Float16 v8h  __attribute__((ext_vector_type(8)));
typedef _Float16 v16h __attribute__((ext_vector_type(16)));
typedef float    v8f  __attribute__((ext_vector_type(8)));

__device__ __forceinline__ v16h make_frag(v8h lo, v8h hi) {
    v16h r;
#pragma unroll
    for (int e = 0; e < 8; ++e) { r[e] = lo[e]; r[8 + e] = hi[e]; }
    return r;
}

// A-fragment (16x32 f16): lane m=l&15 holds row m; elements 0..15 = dims kb+hi*16..
// kb+hi*16+15  -> two contiguous 128-bit loads per lane.
__device__ __forceinline__ v16h load_afrag(const _Float16* rowPtr, int laneHi, int kb) {
    v8h lo = *(const v8h*)(rowPtr + kb + laneHi * 16);
    v8h hi = *(const v8h*)(rowPtr + kb + laneHi * 16 + 8);
    return make_frag(lo, hi);
}

// B-fragment (32x16 f16) used as "rows of the transposed operand": lane holds one
// source row, 16 contiguous dims starting at the lane-group offset.
__device__ __forceinline__ v16h load_bfrag(const _Float16* p) {
    v8h lo = *(const v8h*)(p);
    v8h hi = *(const v8h*)(p + 8);
    return make_frag(lo, hi);
}

__device__ __forceinline__ v8f wmma32(v16h a, v16h b, v8f c) {
    return __builtin_amdgcn_wmma_f32_16x16x32_f16(false, a, false, b, (short)0, c,
                                                  false, false);
}

// ---------------- f32 -> f16 convert (with scale) ----------------
__global__ void cvt_f32_f16(const float* __restrict__ in, _Float16* __restrict__ out,
                            float scale, int n) {
    int i = blockIdx.x * blockDim.x + threadIdx.x;
    if (i < n) out[i] = (_Float16)(in[i] * scale);
}

// ---------------- Weff = W + 2 * B @ A  (f16 out) ----------------
__global__ void build_weff(const float* __restrict__ W, const float* __restrict__ A,
                           const float* __restrict__ Bm, _Float16* __restrict__ out) {
    int i = blockIdx.x * blockDim.x + threadIdx.x;  // over EE*EE
    int n = i >> 10, k = i & (EE - 1);
    float s = W[i];
#pragma unroll
    for (int r = 0; r < RR; ++r)
        s += LORA_SCALE * Bm[n * RR + r] * A[r * EE + k];
    out[i] = (_Float16)s;
}

// ---------------- WMMA GEMM:  out[m,n] = sum_k Am[m,k]*Wm[n,k] + bias[n] ----------
// Each wave computes a 32x64 tile: 8 f32 accumulators (64 VGPRs -> no spills),
// 8 wmma per 32-wide K step.  Named scalar base pointers + induction indexing so
// address-space inference yields global_load_b128 with immediate offsets.
template <bool F16OUT>
__global__ void __launch_bounds__(128)
gemm32x64(const _Float16* __restrict__ Am, const _Float16* __restrict__ Wm,
          const float* __restrict__ bias, void* __restrict__ outp, int M) {
    int wid    = blockIdx.x * (blockDim.x >> 5) + (threadIdx.x >> 5);
    int lane   = threadIdx.x & 31;
    int laneLo = lane & 15, laneHi = lane >> 4;
    const int ntb = EE / 64;                 // 16 column tiles
    int m0 = (wid / ntb) * 32;
    int n0 = (wid % ntb) * 64;
    if (m0 >= M) return;

    // fixed per-lane base pointers (laneHi folded in); index by K inside the loop
    const _Float16* a0 = Am + (size_t)(m0 + laneLo) * EE + laneHi * 16;
    const _Float16* a1 = a0 + (size_t)16 * EE;
    const _Float16* b0 = Wm + (size_t)(n0 + laneLo) * EE + laneHi * 16;
    const _Float16* b1 = b0 + (size_t)16 * EE;
    const _Float16* b2 = b0 + (size_t)32 * EE;
    const _Float16* b3 = b0 + (size_t)48 * EE;

    v8f acc[2][4] = {};
#pragma unroll 4
    for (int kb = 0; kb < EE; kb += 32) {
        v16h af[2], bf[4];
        af[0] = make_frag(*(const v8h*)(a0 + kb), *(const v8h*)(a0 + kb + 8));
        af[1] = make_frag(*(const v8h*)(a1 + kb), *(const v8h*)(a1 + kb + 8));
        bf[0] = make_frag(*(const v8h*)(b0 + kb), *(const v8h*)(b0 + kb + 8));
        bf[1] = make_frag(*(const v8h*)(b1 + kb), *(const v8h*)(b1 + kb + 8));
        bf[2] = make_frag(*(const v8h*)(b2 + kb), *(const v8h*)(b2 + kb + 8));
        bf[3] = make_frag(*(const v8h*)(b3 + kb), *(const v8h*)(b3 + kb + 8));
#pragma unroll
        for (int i = 0; i < 2; ++i)
#pragma unroll
            for (int j = 0; j < 4; ++j)
                acc[i][j] = wmma32(af[i], bf[j], acc[i][j]);
    }

#pragma unroll
    for (int j = 0; j < 4; ++j) {
        int col = n0 + 16 * j + laneLo;
        float bc = bias[col];
#pragma unroll
        for (int i = 0; i < 2; ++i) {
#pragma unroll
            for (int r = 0; r < 8; ++r) {
                int row = m0 + 16 * i + r + 8 * laneHi;   // D-frag row mapping
                float v = acc[i][j][r] + bc;
                if (F16OUT) ((_Float16*)outp)[(size_t)row * EE + col] = (_Float16)v;
                else        ((float*)outp)[(size_t)row * EE + col] = v;
            }
        }
    }
}

// ---------------- stable counting sort (matches stable argsort) ----------------
// grid = 6 blocks: level = bid/2, batch = bid%2.  sidx[level][batch][j] = orig pos.
__global__ void bucket_sort(const int* __restrict__ wb0, const int* __restrict__ wb1,
                            const int* __restrict__ wb2, int* __restrict__ sidx) {
    __shared__ int cnt[256];
    __shared__ int off[256];
    int lvl = blockIdx.x >> 1, b = blockIdx.x & 1;
    const int* keys = ((lvl == 0) ? wb0 : (lvl == 1) ? wb1 : wb2) + b * NN;
    int* out = sidx + (lvl * 2 + b) * NN;
    for (int i = threadIdx.x; i < 256; i += blockDim.x) cnt[i] = 0;
    __syncthreads();
    for (int i = threadIdx.x; i < NN; i += blockDim.x) atomicAdd(&cnt[keys[i]], 1);
    __syncthreads();
    if (threadIdx.x == 0) {
        int s = 0;
        for (int i = 0; i < 256; ++i) { off[i] = s; s += cnt[i]; }
        for (int n = 0; n < NN; ++n) {      // stable scatter
            int k = keys[n];
            out[off[k]++] = n;
        }
    }
}

// ---------------- chunked attention, one level ----------------
// grid = BH * nc blocks, blockDim = (C/16)*32.  Wave w handles query rows
// chunk*C + 16w .. +15 (sorted order).  Flash-style online softmax over 32-key steps.
__global__ void attn_chunked(const _Float16* __restrict__ qbuf,
                             const _Float16* __restrict__ kbuf,
                             const _Float16* __restrict__ vbuf,
                             const int* __restrict__ sidx,   // [2][NN] for this level
                             float* __restrict__ out_attn,
                             int C, int nc, int accumulate) {
    __shared__ __align__(16) _Float16 Klds[32][64];       // K rows (row-major)
    __shared__ __align__(16) _Float16 Vt[64][32];         // V transposed (dim-major)
    __shared__ __align__(16) _Float16 Plds[16][16][32];   // per-wave P bounce

    int bh = blockIdx.x / nc, chunk = blockIdx.x % nc;
    int b = bh >> 4, h = bh & 15;
    int wave = threadIdx.x >> 5, lane = threadIdx.x & 31;
    int laneLo = lane & 15, laneHi = lane >> 4;
    const int* sb = sidx + b * NN;

    // Q A-fragments for this wave's 16 rows (gathered through sort permutation)
    int qrow = sb[chunk * C + wave * 16 + laneLo];
    const _Float16* qp = qbuf + ((size_t)(b * NN + qrow)) * EE + h * HD;
    v16h qa0 = load_afrag(qp, laneHi, 0);
    v16h qa1 = load_afrag(qp, laneHi, 32);

    float mi[8], li[8];
#pragma unroll
    for (int r = 0; r < 8; ++r) { mi[r] = -3.0e38f; li[r] = 0.0f; }
    v8f O[4] = {};

    int jstart = (chunk == 0) ? C : 0;          // chunk 0: look-back fully masked
    int nkeys  = 2 * C - jstart;
    int nsteps = (nkeys + 31) >> 5;

    for (int s = 0; s < nsteps; ++s) {
        int j0 = jstart + s * 32;
        __syncthreads();
        // cooperative stage of 32 sorted K rows + transposed V rows
        for (int idx = threadIdx.x; idx < 256; idx += blockDim.x) {
            int kk = idx >> 3;
            int d8 = (idx & 7) * 8;
            int kpos = chunk * C - C + j0 + kk;
            if (kpos >= NN) kpos = NN - 1;       // masked tail (C=16, chunk 0)
            int krow = sb[kpos];
            size_t base = ((size_t)(b * NN + krow)) * EE + h * HD + d8;
            *(v8h*)&Klds[kk][d8] = *(const v8h*)(kbuf + base);
            v8h vv = *(const v8h*)(vbuf + base);
#pragma unroll
            for (int e = 0; e < 8; ++e) Vt[d8 + e][kk] = vv[e];
        }
        __syncthreads();

        // S = (Q @ K^T) / 8 : two 16x16 tiles, K-dim 64 = 2 wmma each
        v8f S[2];
#pragma unroll
        for (int t = 0; t < 2; ++t) {
            v16h kb0 = load_bfrag(&Klds[16 * t + laneLo][laneHi * 16]);
            v16h kb1 = load_bfrag(&Klds[16 * t + laneLo][32 + laneHi * 16]);
            v8f z = {};
            z = wmma32(qa0, kb0, z);
            z = wmma32(qa1, kb1, z);
#pragma unroll
            for (int r = 0; r < 8; ++r) z[r] *= 0.125f;
            if (j0 + 16 * t >= 2 * C) {          // half-step mask (C=16, chunk 0)
#pragma unroll
                for (int r = 0; r < 8; ++r) z[r] = -3.0e38f;
            }
            S[t] = z;
        }

        // online softmax: row reductions across 16-lane halves (rows follow lanes)
        float mx[8];
#pragma unroll
        for (int r = 0; r < 8; ++r) {
            float v = fmaxf(S[0][r], S[1][r]);
            v = fmaxf(v, __shfl_xor(v, 1, 32));
            v = fmaxf(v, __shfl_xor(v, 2, 32));
            v = fmaxf(v, __shfl_xor(v, 4, 32));
            v = fmaxf(v, __shfl_xor(v, 8, 32));
            mx[r] = v;
        }
        float rs[8];
#pragma unroll
        for (int r = 0; r < 8; ++r) {
            float mn = fmaxf(mi[r], mx[r]);
            float sc = __expf(mi[r] - mn);
            mi[r] = mn;
            li[r] *= sc;
            float p0 = __expf(S[0][r] - mn);
            float p1 = __expf(S[1][r] - mn);
            S[0][r] = p0; S[1][r] = p1;
            rs[r] = p0 + p1;
#pragma unroll
            for (int t = 0; t < 4; ++t) O[t][r] *= sc;
        }
#pragma unroll
        for (int r = 0; r < 8; ++r) {
            float v = rs[r];
            v += __shfl_xor(v, 1, 32);
            v += __shfl_xor(v, 2, 32);
            v += __shfl_xor(v, 4, 32);
            v += __shfl_xor(v, 8, 32);
            li[r] += v;
        }

        // D-layout P -> LDS -> A-fragment (transpose via wave-private LDS)
#pragma unroll
        for (int r = 0; r < 8; ++r) {
            Plds[wave][r + 8 * laneHi][laneLo]      = (_Float16)S[0][r];
            Plds[wave][r + 8 * laneHi][16 + laneLo] = (_Float16)S[1][r];
        }
        __builtin_amdgcn_wave_barrier();   // DS is in-order per wave; stop reordering
        v16h pa;
        {
            v8h lo = *(const v8h*)&Plds[wave][laneLo][laneHi * 16];
            v8h hi = *(const v8h*)&Plds[wave][laneLo][laneHi * 16 + 8];
            pa = make_frag(lo, hi);
        }
        __builtin_amdgcn_wave_barrier();

        // O += P @ V : 4 output dim-tiles, B-frags from transposed V in LDS
#pragma unroll
        for (int t = 0; t < 4; ++t) {
            v16h vb = load_bfrag(&Vt[16 * t + laneLo][laneHi * 16]);
            O[t] = wmma32(pa, vb, O[t]);
        }
    }

    // normalize + scatter back through the permutation (out[sidx[j]] = out_sorted[j])
#pragma unroll
    for (int r = 0; r < 8; ++r) {
        int p = sb[chunk * C + wave * 16 + r + 8 * laneHi];
        float inv = 1.0f / li[r];
#pragma unroll
        for (int t = 0; t < 4; ++t) {
            size_t addr = ((size_t)(b * NN + p)) * EE + h * HD + 16 * t + laneLo;
            float val = O[t][r] * inv;
            if (accumulate) out_attn[addr] += val;
            else            out_attn[addr]  = val;
        }
    }
}

extern "C" void kernel_launch(void* const* d_in, const int* in_sizes, int n_in,
                              void* d_out, int out_size, void* d_ws, size_t ws_size,
                              hipStream_t stream) {
    (void)in_sizes; (void)n_in; (void)out_size; (void)ws_size;
    const float* x   = (const float*)d_in[0];
    const int*   wb0 = (const int*)d_in[1];   // coarse -> C=256
    const int*   wb1 = (const int*)d_in[2];   // mid    -> C=64
    const int*   wb2 = (const int*)d_in[3];   // fine   -> C=16
    const float* Wp[4], *bp[4], *Ap[4], *Bp[4];
    for (int i = 0; i < 4; ++i) {             // q, k, v, o
        Wp[i] = (const float*)d_in[4 + 4 * i + 0];
        bp[i] = (const float*)d_in[4 + 4 * i + 1];
        Ap[i] = (const float*)d_in[4 + 4 * i + 2];
        Bp[i] = (const float*)d_in[4 + 4 * i + 3];
    }

    char* ws = (char*)d_ws;
    size_t off = 0;
    auto carve = [&](size_t bytes) {
        void* p = ws + off;
        off = (off + bytes + 255) & ~(size_t)255;
        return p;
    };
    _Float16* Xh   = (_Float16*)carve((size_t)MN * EE * 2);   // x (f16); reused for Oh
    _Float16* Wh[4];
    for (int i = 0; i < 4; ++i) Wh[i] = (_Float16*)carve((size_t)EE * EE * 2);
    _Float16* qb = (_Float16*)carve((size_t)MN * EE * 2);
    _Float16* kb = (_Float16*)carve((size_t)MN * EE * 2);
    _Float16* vb = (_Float16*)carve((size_t)MN * EE * 2);
    int*   sidx = (int*)carve((size_t)3 * 2 * NN * 4);
    float* oat  = (float*)carve((size_t)MN * EE * 4);

    const int nElem = MN * EE;                // 8388608
    cvt_f32_f16<<<(nElem + 255) / 256, 256, 0, stream>>>(x, Xh, 1.0f, nElem);
    for (int i = 0; i < 4; ++i)
        build_weff<<<(EE * EE) / 256, 256, 0, stream>>>(Wp[i], Ap[i], Bp[i], Wh[i]);

    // q/k/v projections (f16 out):  (8192/32)*(1024/64) = 4096 wave tiles, 4 waves/block
    gemm32x64<true><<<1024, 128, 0, stream>>>(Xh, Wh[0], bp[0], qb, MN);
    gemm32x64<true><<<1024, 128, 0, stream>>>(Xh, Wh[1], bp[1], kb, MN);
    gemm32x64<true><<<1024, 128, 0, stream>>>(Xh, Wh[2], bp[2], vb, MN);

    bucket_sort<<<6, 256, 0, stream>>>(wb0, wb1, wb2, sidx);

    // three attention levels: level 0 initializes, 1 & 2 accumulate (serialized)
    attn_chunked<<<32 * 16,  512, 0, stream>>>(qb, kb, vb, sidx + 0 * 2 * NN, oat, 256, 16,  0);
    attn_chunked<<<32 * 64,  128, 0, stream>>>(qb, kb, vb, sidx + 1 * 2 * NN, oat, 64,  64,  1);
    attn_chunked<<<32 * 256,  32, 0, stream>>>(qb, kb, vb, sidx + 2 * 2 * NN, oat, 16,  256, 1);

    // average of 3 levels folded into the f16 convert; o-projection -> d_out (f32)
    cvt_f32_f16<<<(nElem + 255) / 256, 256, 0, stream>>>(oat, Xh, 1.0f / 3.0f, nElem);
    gemm32x64<false><<<1024, 128, 0, stream>>>(Xh, Wh[3], bp[3], (float*)d_out, MN);
}
